// Attention_54382875902242
// MI455X (gfx1250) — matrix operand order repaired and checked by reading, not compiled, not run on hardware
//
#include <hip/hip_runtime.h>
#include <math.h>

#define HW  16384
#define WD  128
#define HT  128
#define BN  8

typedef __attribute__((ext_vector_type(16))) __bf16 v16bf;
typedef __attribute__((ext_vector_type(8)))  float  v8f;
typedef __attribute__((ext_vector_type(4)))  float  f4;

__device__ __forceinline__ float gelu_exact(float z) {
  return 0.5f * z * (1.0f + erff(z * 0.70710678118654752f));
}
__device__ __forceinline__ float sigmoidf_(float z) {
  return 1.0f / (1.0f + expf(-z));
}

// ---------------------------------------------------------------------------
// WMMA GEMM:  Y[b] (M x HW) = W (M x K) @ X[b] (K x HW) (+bias)
// XFIN==1: X'[k][n] = gelu(aux[b*K+k] * X[k][n]);  XFOUT==1: relu output.
// 128 threads = 4 wave32; 64(M) x 128(N) tile; bf16 inputs, f32 accumulate.
// Double-buffered LDS + software pipeline: global loads for tile k0+32 are
// issued before the 8 WMMAs of tile k0; one barrier per K-step.
// ---------------------------------------------------------------------------
template<int XFIN, int XFOUT>
__global__ __launch_bounds__(128)
void gemm_wmma(const float* __restrict__ Wm, const float* __restrict__ X,
               const float* __restrict__ bias, const float* __restrict__ aux,
               float* __restrict__ Y, int M, int K,
               long long xbs, long long ybs)
{
  __shared__ __bf16 Bs[2][128][40];       // [buf][n][k], padded stride (20 KB)
  const int b    = blockIdx.z;
  const float* Xb = X + (long long)b * xbs;
  float*       Yb = Y + (long long)b * ybs;
  const float* auxb = (XFIN == 1) ? (aux + (long long)b * K) : nullptr;
  const int n0   = blockIdx.x * 128;
  const int m0   = blockIdx.y * 64;
  const int tid  = threadIdx.x;
  const int lane = tid & 31;
  const int wv   = tid >> 5;
  const int lm   = lane & 15;
  const int kbA  = (lane < 16) ? 0 : 8;   // A-matrix K split {0-7,16-23}/{8-15,24-31}
  const int kbB  = (lane < 16) ? 0 : 8;   // B-matrix K split {0-7,16-23}/{8-15,24-31}
  const int arow = m0 + wv * 16 + lm;

  // staging descriptors: l = tid + i*128 over 512 pair-blocks,
  // kp = l>>5 (K pair 0..15 -> k = 2*kp, 2*kp+1), n4 = (l&31)*4
  const float* gptr[4];
  unsigned     soff[4];                   // element offset within one buffer
  int          kparr[4];
  #pragma unroll
  for (int i = 0; i < 4; ++i) {
    int l  = tid + i * 128;
    int kp = l >> 5;
    int n4 = (l & 31) * 4;
    kparr[i] = kp;
    gptr[i]  = Xb + (size_t)(2 * kp) * HW + n0 + n4;
    soff[i]  = (unsigned)(n4 * 40 + 2 * kp);
  }

  auto load_tile = [&](int kk, f4* va, f4* vb) {
    #pragma unroll
    for (int i = 0; i < 4; ++i) {
      const float* g = gptr[i] + (size_t)kk * HW;
      va[i] = *(const f4*)g;
      vb[i] = *(const f4*)(g + HW);
    }
  };
  auto store_tile = [&](int kk, f4* va, f4* vb, int buf) {
    __bf16* base = &Bs[buf][0][0];
    #pragma unroll
    for (int i = 0; i < 4; ++i) {
      if constexpr (XFIN == 1) {
        float s0 = auxb[kk + 2 * kparr[i]];
        float s1 = auxb[kk + 2 * kparr[i] + 1];
        #pragma unroll
        for (int j = 0; j < 4; ++j) {
          va[i][j] = gelu_exact(s0 * va[i][j]);
          vb[i][j] = gelu_exact(s1 * vb[i][j]);
        }
      }
      #pragma unroll
      for (int j = 0; j < 4; ++j) {
        union { __bf16 h[2]; unsigned u; } pk;
        pk.h[0] = (__bf16)va[i][j];       // k = 2*kp
        pk.h[1] = (__bf16)vb[i][j];       // k = 2*kp+1
        *(unsigned*)(base + soff[i] + j * 40) = pk.u;
      }
    }
  };
  auto load_afrag = [&](int kk) -> v16bf {
    v16bf af;
    #pragma unroll
    for (int j = 0; j < 16; ++j) af[j] = (__bf16)0.0f;
    if (arow < M) {
      const float* wr = Wm + (size_t)arow * K + kk;
      f4 w0 = *(const f4*)(wr + kbA);
      f4 w1 = *(const f4*)(wr + kbA + 4);
      f4 w2 = *(const f4*)(wr + 16 + kbA);
      f4 w3 = *(const f4*)(wr + 16 + kbA + 4);
      #pragma unroll
      for (int j = 0; j < 4; ++j) {
        af[j]      = (__bf16)w0[j];
        af[j + 4]  = (__bf16)w1[j];
        af[j + 8]  = (__bf16)w2[j];
        af[j + 12] = (__bf16)w3[j];
      }
    }
    return af;
  };

  v8f acc[8];
  #pragma unroll
  for (int t = 0; t < 8; ++t)
    #pragma unroll
    for (int j = 0; j < 8; ++j) acc[t][j] = 0.0f;

  // prologue: stage tile 0 into buffer 0, preload A fragment 0
  {
    f4 va[4], vb[4];
    load_tile(0, va, vb);
    store_tile(0, va, vb, 0);
  }
  v16bf afc = load_afrag(0);
  __syncthreads();

  int pbuf = 0;
  for (int k0 = 0; k0 < K; k0 += 32) {
    const int  k1   = k0 + 32;
    const bool have = (k1 < K);

    // issue next tile's global loads before the WMMAs (latency hiding)
    f4 na[4], nb[4];
    if (have) load_tile(k1, na, nb);
    v16bf afn = afc;
    if (have) afn = load_afrag(k1);

    #pragma unroll
    for (int t = 0; t < 8; ++t) {
      v16bf bf;
      int nn = t * 16 + lm;
      #pragma unroll
      for (int j = 0; j < 16; ++j) bf[j] = Bs[pbuf][nn][kbB + j + ((j & 8) ? 8 : 0)];
      acc[t] = __builtin_amdgcn_wmma_f32_16x16x32_bf16(false, afc, false, bf,
                                                       (short)0, acc[t], false, false);
    }

    if (have) store_tile(k1, na, nb, pbuf ^ 1);
    __syncthreads();
    afc  = afn;
    pbuf ^= 1;
  }

  #pragma unroll
  for (int t = 0; t < 8; ++t) {
    int col = n0 + t * 16 + lm;
    #pragma unroll
    for (int r = 0; r < 8; ++r) {
      int row = m0 + wv * 16 + r + ((lane < 16) ? 0 : 8);
      if (row < M) {
        float v = acc[t][r];
        if (bias) v += bias[row];
        if constexpr (XFOUT == 1) v = fmaxf(v, 0.0f);
        Yb[(size_t)row * HW + col] = v;
      }
    }
  }
}

// ---------------------------------------------------------------------------
// depthwise / grouped 3x3 conv, pad 1.  in_ch = oc / ratio.
// ---------------------------------------------------------------------------
__global__ void dwconv3x3(const float* __restrict__ in, float* __restrict__ out,
                          const float* __restrict__ w, const float* __restrict__ bias,
                          int Cout, int ratio, long long inBS, long long outBS)
{
  long long idx = (long long)blockIdx.x * 256 + threadIdx.x;
  int p  = (int)(idx & (HW - 1));
  int oc = (int)((idx >> 14) % Cout);
  int b  = (int)(idx / ((long long)HW * Cout));
  int yy = p >> 7, xx = p & 127;
  const float* ib = in + (long long)b * inBS + (long long)(oc / ratio) * HW;
  const float* wk = w + oc * 9;
  float acc = bias ? bias[oc] : 0.0f;
  #pragma unroll
  for (int ky = 0; ky < 3; ++ky) {
    int iy = yy + ky - 1;
    if (iy < 0 || iy >= HT) continue;
    #pragma unroll
    for (int kx = 0; kx < 3; ++kx) {
      int ix = xx + kx - 1;
      if (ix < 0 || ix >= WD) continue;
      acc += wk[ky * 3 + kx] * ib[iy * WD + ix];
    }
  }
  out[(long long)b * outBS + (long long)oc * HW + p] = acc;
}

// ---------------------------------------------------------------------------
// per-channel mean over HW (block per [b*C + c] slab, contiguous layout)
// ---------------------------------------------------------------------------
__global__ void chan_mean(const float* __restrict__ in, float* __restrict__ out)
{
  __shared__ float red[256];
  const float* p = in + (size_t)blockIdx.x * HW;
  float s = 0.f;
  for (int i = threadIdx.x; i < HW; i += 256) s += p[i];
  red[threadIdx.x] = s; __syncthreads();
  for (int st = 128; st > 0; st >>= 1) {
    if (threadIdx.x < st) red[threadIdx.x] += red[threadIdx.x + st];
    __syncthreads();
  }
  if (threadIdx.x == 0) out[blockIdx.x] = red[0] * (1.0f / HW);
}

// per-channel L2 norm of qkvd channels 0..255 (q then k), batch stride 384*HW
__global__ void chan_l2norm(const float* __restrict__ qkvd, float* __restrict__ out)
{
  __shared__ float red[256];
  int b = blockIdx.x >> 8;
  int c = blockIdx.x & 255;
  const float* p = qkvd + (size_t)b * 384 * HW + (size_t)c * HW;
  float s = 0.f;
  for (int i = threadIdx.x; i < HW; i += 256) { float v = p[i]; s += v * v; }
  red[threadIdx.x] = s; __syncthreads();
  for (int st = 128; st > 0; st >>= 1) {
    if (threadIdx.x < st) red[threadIdx.x] += red[threadIdx.x + st];
    __syncthreads();
  }
  if (threadIdx.x == 0) out[blockIdx.x] = sqrtf(red[0]);
}

// softmax over 256 channel means, one block per batch
__global__ void softmax256(const float* __restrict__ m, float* __restrict__ t1)
{
  __shared__ float red[256];
  int b = blockIdx.x;
  float v = m[b * 256 + threadIdx.x];
  red[threadIdx.x] = v; __syncthreads();
  for (int st = 128; st > 0; st >>= 1) {
    if (threadIdx.x < st) red[threadIdx.x] = fmaxf(red[threadIdx.x], red[threadIdx.x + st]);
    __syncthreads();
  }
  float mx = red[0]; __syncthreads();
  float e = expf(v - mx);
  red[threadIdx.x] = e; __syncthreads();
  for (int st = 128; st > 0; st >>= 1) {
    if (threadIdx.x < st) red[threadIdx.x] += red[threadIdx.x + st];
    __syncthreads();
  }
  t1[b * 256 + threadIdx.x] = e / red[0];
}

// gate second conv + sigmoid + per-block partial sum (deterministic, no atomics)
__global__ void gate_partial(const float* __restrict__ g1, const float* __restrict__ w2,
                             const float* __restrict__ b2, float* __restrict__ part)
{
  __shared__ float red[256];
  int b = blockIdx.y;
  int p = blockIdx.x * 256 + threadIdx.x;
  const float* gb = g1 + (size_t)b * 64 * HW + p;
  float s = b2[0];
  for (int c = 0; c < 64; ++c) s += w2[c] * gb[(size_t)c * HW];
  red[threadIdx.x] = sigmoidf_(s); __syncthreads();
  for (int st = 128; st > 0; st >>= 1) {
    if (threadIdx.x < st) red[threadIdx.x] += red[threadIdx.x + st];
    __syncthreads();
  }
  if (threadIdx.x == 0) part[b * 64 + blockIdx.x] = red[0];
}

__global__ void gate_final(const float* __restrict__ part, float* __restrict__ dynk)
{
  __shared__ float red[256];
  float s = part[threadIdx.x] + part[threadIdx.x + 256];
  red[threadIdx.x] = s; __syncthreads();
  for (int st = 128; st > 0; st >>= 1) {
    if (threadIdx.x < st) red[threadIdx.x] += red[threadIdx.x + st];
    __syncthreads();
  }
  if (threadIdx.x == 0) {
    float gate = red[0] / (float)(BN * HW);
    gate = fminf(fmaxf(gate, 0.1f), 1.0f);
    float dk = floorf(16.0f * gate);
    dynk[0] = fminf(fmaxf(dk, 1.0f), 16.0f);
  }
}

// Q·K^T per (b,h), normalized by L2 norms, * temperature[h]
__global__ __launch_bounds__(256)
void attn_qk(const float* __restrict__ qkvd, const float* __restrict__ norms,
             const float* __restrict__ temp, float* __restrict__ attn)
{
  __shared__ float qs[16][132];
  __shared__ float ks[16][132];
  int b = blockIdx.x >> 3, h = blockIdx.x & 7;
  int tid = threadIdx.x, c = tid >> 4, d = tid & 15;
  const float* qb = qkvd + (size_t)b * 384 * HW + (size_t)(h * 16) * HW;
  const float* kp = qkvd + (size_t)b * 384 * HW + (size_t)(128 + h * 16) * HW;
  float acc = 0.f;
  for (int n0 = 0; n0 < HW; n0 += 128) {
    #pragma unroll
    for (int i = 0; i < 8; ++i) {
      int l = tid + i * 256;
      int r = l >> 7, j = l & 127;
      qs[r][j] = qb[(size_t)r * HW + n0 + j];
      ks[r][j] = kp[(size_t)r * HW + n0 + j];
    }
    __syncthreads();
    #pragma unroll 8
    for (int j = 0; j < 128; ++j) acc += qs[c][j] * ks[d][j];
    __syncthreads();
  }
  float nq = fmaxf(norms[b * 256 + h * 16 + c], 1e-8f);
  float nk = fmaxf(norms[b * 256 + 128 + h * 16 + d], 1e-8f);
  attn[(size_t)blockIdx.x * 256 + tid] = acc / (nq * nk) * temp[h];
}

// rank-based top-k mask (== argsort(argsort(-a)) < k, stable ties) + row softmax
__global__ void mask_softmax(float* __restrict__ attn, const float* __restrict__ dynkf)
{
  __shared__ float A[16][17];
  __shared__ float Bm[16][17];
  int tid = threadIdx.x, c = tid >> 4, d = tid & 15;
  float* base = attn + (size_t)blockIdx.x * 256;
  A[c][d] = base[tid];
  __syncthreads();
  int Kk = (int)(dynkf[0] + 0.5f);
  Kk = (Kk < 1) ? 1 : ((Kk > 16) ? 16 : Kk);
  float v = A[c][d];
  int rank = 0;
  #pragma unroll
  for (int j = 0; j < 16; ++j) {
    float o = A[c][j];
    rank += (o > v) || (o == v && j < d);
  }
  float val = (rank < Kk) ? fminf(fmaxf(v, -1e8f), 1e8f) : -1e8f;
  Bm[c][d] = val; __syncthreads();
  float mx = -INFINITY;
  #pragma unroll
  for (int j = 0; j < 16; ++j) mx = fmaxf(mx, Bm[c][j]);
  float e = expf(val - mx);
  A[c][d] = e; __syncthreads();
  float s = 0.f;
  #pragma unroll
  for (int j = 0; j < 16; ++j) s += A[c][j];
  base[tid] = e / s;
}

// out_att = (attn @ V) * (a1+a2+a3+a4)
__global__ void attn_ov(const float* __restrict__ qkvd, const float* __restrict__ attn,
                        const float* __restrict__ a1, const float* __restrict__ a2,
                        const float* __restrict__ a3, const float* __restrict__ a4,
                        float* __restrict__ outatt)
{
  __shared__ float A[256];
  int b = blockIdx.z, h = blockIdx.y;
  int p = blockIdx.x * 256 + threadIdx.x;
  A[threadIdx.x] = attn[(size_t)(b * 8 + h) * 256 + threadIdx.x];
  __syncthreads();
  float scale = a1[0] + a2[0] + a3[0] + a4[0];
  const float* vb = qkvd + (size_t)b * 384 * HW + (size_t)(256 + h * 16) * HW + p;
  float vv[16];
  #pragma unroll
  for (int d = 0; d < 16; ++d) vv[d] = vb[(size_t)d * HW];
  float* ob = outatt + (size_t)b * 128 * HW + (size_t)(h * 16) * HW + p;
  #pragma unroll
  for (int c = 0; c < 16; ++c) {
    float s = 0.f;
    #pragma unroll
    for (int d = 0; d < 16; ++d) s += A[c * 16 + d] * vv[d];
    ob[(size_t)c * HW] = s * scale;
  }
}

// channel interaction (tiny): conv 128->32, GN(32), gelu, conv 32->128, sigmoid
__global__ __launch_bounds__(128)
void chan_inter(const float* __restrict__ cmean,
                const float* __restrict__ w1, const float* __restrict__ b1,
                const float* __restrict__ gw, const float* __restrict__ gb,
                const float* __restrict__ w2, const float* __restrict__ b2,
                float* __restrict__ cmsig)
{
  __shared__ float cms[128];
  __shared__ float z[32];
  __shared__ float stat[2];
  int b = blockIdx.x, t = threadIdx.x;
  cms[t] = cmean[b * 128 + t];
  __syncthreads();
  if (t < 32) {
    float s = b1[t];
    for (int i = 0; i < 128; ++i) s += w1[t * 128 + i] * cms[i];
    z[t] = s;
  }
  __syncthreads();
  if (t == 0) {
    float m = 0.f; for (int j = 0; j < 32; ++j) m += z[j]; m *= (1.f / 32.f);
    float v = 0.f; for (int j = 0; j < 32; ++j) { float d = z[j] - m; v += d * d; }
    v *= (1.f / 32.f);
    stat[0] = m; stat[1] = rsqrtf(v + 1e-5f);
  }
  __syncthreads();
  if (t < 32) z[t] = gelu_exact((z[t] - stat[0]) * stat[1] * gw[t] + gb[t]);
  __syncthreads();
  float s = b2[t];
  for (int j = 0; j < 32; ++j) s += w2[t * 32 + j] * z[j];
  cmsig[b * 128 + t] = sigmoidf_(s);
}

// GroupNorm(1) stats over sm0[b] (16 x HW)
__global__ void gn_stats(const float* __restrict__ sm0, float* __restrict__ gn)
{
  __shared__ float rs[256], rq[256];
  int b = blockIdx.x;
  const float* p = sm0 + (size_t)b * 16 * HW;
  float s = 0.f, q = 0.f;
  for (int i = threadIdx.x; i < 16 * HW; i += 256) { float v = p[i]; s += v; q += v * v; }
  rs[threadIdx.x] = s; rq[threadIdx.x] = q; __syncthreads();
  for (int st = 128; st > 0; st >>= 1) {
    if (threadIdx.x < st) { rs[threadIdx.x] += rs[threadIdx.x + st]; rq[threadIdx.x] += rq[threadIdx.x + st]; }
    __syncthreads();
  }
  if (threadIdx.x == 0) {
    float n = 16.f * HW;
    float m = rs[0] / n;
    float var = rq[0] / n - m * m;
    gn[b * 2] = m; gn[b * 2 + 1] = rsqrtf(fmaxf(var, 0.f) + 1e-5f);
  }
}

// attened = out_att * sigmoid(si(y_d)); conv_x = y_d * sigmoid(cm) -> concat buf
__global__ void fuse_concat(const float* __restrict__ outatt, const float* __restrict__ yd,
                            const float* __restrict__ sm0, const float* __restrict__ gn,
                            const float* __restrict__ sgw, const float* __restrict__ sgb,
                            const float* __restrict__ sw2, const float* __restrict__ sb2,
                            const float* __restrict__ cmsig, float* __restrict__ concat)
{
  __shared__ float cms[128];
  int b = blockIdx.y;
  int p = blockIdx.x * 256 + threadIdx.x;
  if (threadIdx.x < 128) cms[threadIdx.x] = cmsig[b * 128 + threadIdx.x];
  __syncthreads();
  float m = gn[b * 2], inv = gn[b * 2 + 1];
  float s = sb2[0];
  #pragma unroll
  for (int c = 0; c < 16; ++c) {
    float x = sm0[(size_t)b * 16 * HW + (size_t)c * HW + p];
    x = (x - m) * inv * sgw[c] + sgb[c];
    s += sw2[c] * gelu_exact(x);
  }
  float sig = sigmoidf_(s);
  const float* oa = outatt + (size_t)b * 128 * HW + p;
  const float* yv = yd + (size_t)b * 128 * HW + p;
  float* cc = concat + (size_t)b * 256 * HW + p;
  for (int c = 0; c < 128; ++c) cc[(size_t)c * HW] = oa[(size_t)c * HW] * sig;
  for (int c = 0; c < 128; ++c) cc[(size_t)(128 + c) * HW] = yv[(size_t)c * HW] * cms[c];
}

// ---------------------------------------------------------------------------
// workspace layout (float offsets); regions reused once dead
// ---------------------------------------------------------------------------
static const size_t O_YX     = 0;            // 33.55M  yx (y | xh)
static const size_t O_T      = 33554432;     // 33.55M  spr dw out
static const size_t O_T2     = 67108864;     // 33.55M  spr pw out
static const size_t O_YD     = 100663296;    // 16.78M  y_d (long-lived)
static const size_t O_QKV0   = 33554432;     // 50.33M  over t/t2 (dead)
static const size_t O_QKVD   = 117440512;    // 50.33M  post-dw qkv
static const size_t O_OUTATT = 0;            // 16.78M  over yx (dead)
static const size_t O_SM0    = 16777216;     //  2.10M
static const size_t O_GBUF   = 117440512;    //  8.39M  temporal reuse w/ qkvd
static const size_t O_CONCAT = 117440512;    // 33.55M  over qkvd (dead)
static const size_t O_SMALL  = 167772160;
static const size_t O_GPART  = O_SMALL;          // 512
static const size_t O_DYNK   = O_SMALL + 1024;   // 1
static const size_t O_MBUF   = O_SMALL + 2048;   // 2048
static const size_t O_T1     = O_SMALL + 4096;   // 2048
static const size_t O_NORMS  = O_SMALL + 6144;   // 2048
static const size_t O_ATTN   = O_SMALL + 8192;   // 16384
static const size_t O_CMEAN  = O_SMALL + 24576;  // 1024
static const size_t O_CMSIG  = O_SMALL + 25600;  // 1024
static const size_t O_GN     = O_SMALL + 26624;  // 16

extern "C" void kernel_launch(void* const* d_in, const int* in_sizes, int n_in,
                              void* d_out, int out_size, void* d_ws, size_t ws_size,
                              hipStream_t stream)
{
  (void)in_sizes; (void)n_in; (void)out_size; (void)ws_size;
  const float* x       = (const float*)d_in[0];
  const float* temp    = (const float*)d_in[1];
  const float* w_lin0  = (const float*)d_in[2];
  const float* b_lin0  = (const float*)d_in[3];
  const float* spr_wdw = (const float*)d_in[4];
  const float* spr_bdw = (const float*)d_in[5];
  const float* spr_wpw = (const float*)d_in[6];
  const float* spr_bpw = (const float*)d_in[7];
  const float* spr_w1  = (const float*)d_in[8];
  const float* spr_b1  = (const float*)d_in[9];
  const float* w_qkv   = (const float*)d_in[10];
  const float* w_dwqkv = (const float*)d_in[11];
  const float* w_proj  = (const float*)d_in[12];
  const float* a1      = (const float*)d_in[13];
  const float* a2      = (const float*)d_in[14];
  const float* a3      = (const float*)d_in[15];
  const float* a4      = (const float*)d_in[16];
  const float* ci_w1   = (const float*)d_in[17];
  const float* ci_b1   = (const float*)d_in[18];
  const float* ci_gw   = (const float*)d_in[19];
  const float* ci_gb   = (const float*)d_in[20];
  const float* ci_w2   = (const float*)d_in[21];
  const float* ci_b2   = (const float*)d_in[22];
  const float* si_w1   = (const float*)d_in[23];
  const float* si_b1   = (const float*)d_in[24];
  const float* si_gw   = (const float*)d_in[25];
  const float* si_gb   = (const float*)d_in[26];
  const float* si_w2   = (const float*)d_in[27];
  const float* si_b2   = (const float*)d_in[28];
  const float* g_w1    = (const float*)d_in[29];
  const float* g_b1    = (const float*)d_in[30];
  const float* g_w2    = (const float*)d_in[31];
  const float* g_b2    = (const float*)d_in[32];

  float* ws     = (float*)d_ws;
  float* yx     = ws + O_YX;
  float* t      = ws + O_T;
  float* t2     = ws + O_T2;
  float* yd     = ws + O_YD;
  float* qkv0   = ws + O_QKV0;
  float* qkvd   = ws + O_QKVD;
  float* outatt = ws + O_OUTATT;
  float* sm0    = ws + O_SM0;
  float* gbuf   = ws + O_GBUF;
  float* concat = ws + O_CONCAT;
  float* gpart  = ws + O_GPART;
  float* dynk   = ws + O_DYNK;
  float* mbuf   = ws + O_MBUF;
  float* t1     = ws + O_T1;
  float* norms  = ws + O_NORMS;
  float* attn   = ws + O_ATTN;
  float* cmean  = ws + O_CMEAN;
  float* cmsig  = ws + O_CMSIG;
  float* gn     = ws + O_GN;
  float* outp   = (float*)d_out;

  // 1: lin0 (256x256)
  gemm_wmma<0,0><<<dim3(128,4,8),128,0,stream>>>(w_lin0, x, b_lin0, nullptr, yx, 256,256, 256LL*HW, 256LL*HW);
  // 2: spr grouped dw 3x3 (128 -> 256, ratio 2)
  dwconv3x3<<<(BN*256*HW)/256,256,0,stream>>>(yx, t, spr_wdw, spr_bdw, 256, 2, 256LL*HW, 256LL*HW);
  // 3: spr pointwise (256x256)
  gemm_wmma<0,0><<<dim3(128,4,8),128,0,stream>>>(spr_wpw, t, spr_bpw, nullptr, t2, 256,256, 256LL*HW, 256LL*HW);
  // 4: channel means + softmax -> t1
  chan_mean<<<BN*256,256,0,stream>>>(t2, mbuf);
  softmax256<<<BN,256,0,stream>>>(mbuf, t1);
  // 5: y_d = spr_w1 @ gelu(t1*t2) + b  (fused input transform)
  gemm_wmma<1,0><<<dim3(128,2,8),128,0,stream>>>(spr_w1, t2, spr_b1, t1, yd, 128,256, 256LL*HW, 128LL*HW);
  // 6: gate branch (relu GEMM + sigmoid dot + deterministic mean) -> dynamic_k
  gemm_wmma<0,1><<<dim3(128,1,8),128,0,stream>>>(g_w1, yx + 128LL*HW, g_b1, nullptr, gbuf, 64,128, 256LL*HW, 64LL*HW);
  gate_partial<<<dim3(64,8),256,0,stream>>>(gbuf, g_w2, g_b2, gpart);
  gate_final<<<1,256,0,stream>>>(gpart, dynk);
  // 7: qkv (384x128), no bias
  gemm_wmma<0,0><<<dim3(128,6,8),128,0,stream>>>(w_qkv, yx + 128LL*HW, nullptr, nullptr, qkv0, 384,128, 256LL*HW, 384LL*HW);
  // 8: depthwise 3x3 on qkv (384 ch), no bias
  dwconv3x3<<<(BN*384*HW)/256,256,0,stream>>>(qkv0, qkvd, w_dwqkv, nullptr, 384, 1, 384LL*HW, 384LL*HW);
  // 9: L2 norms of q,k channels
  chan_l2norm<<<BN*256,256,0,stream>>>(qkvd, norms);
  // 10-12: channel attention
  attn_qk<<<BN*8,256,0,stream>>>(qkvd, norms, temp, attn);
  mask_softmax<<<BN*8,256,0,stream>>>(attn, dynk);
  attn_ov<<<dim3(HW/256,8,BN),256,0,stream>>>(qkvd, attn, a1,a2,a3,a4, outatt);
  // 13: channel interaction
  chan_mean<<<BN*128,256,0,stream>>>(outatt, cmean);
  chan_inter<<<BN,128,0,stream>>>(cmean, ci_w1, ci_b1, ci_gw, ci_gb, ci_w2, ci_b2, cmsig);
  // 14-15: spatial interaction stage 1 + GN stats
  gemm_wmma<0,0><<<dim3(128,1,8),128,0,stream>>>(si_w1, yd, si_b1, nullptr, sm0, 16,128, 128LL*HW, 16LL*HW);
  gn_stats<<<BN,256,0,stream>>>(sm0, gn);
  // 16: fused gating + concat
  fuse_concat<<<dim3(HW/256,BN),256,0,stream>>>(outatt, yd, sm0, gn, si_gw, si_gb, si_w2, si_b2, cmsig, concat);
  // 17: final projection (256x256), no bias -> d_out
  gemm_wmma<0,0><<<dim3(128,4,8),128,0,stream>>>(w_proj, concat, nullptr, nullptr, outp, 256,256, 256LL*HW, 256LL*HW);
}
